// KernelGKN_66812511256913
// MI455X (gfx1250) — compile-verified, matrix operand order checked by reading, not run on hardware
//
#include <hip/hip_runtime.h>
#include <hip/hip_bf16.h>
#include <hip/hip_fp16.h>

typedef __attribute__((ext_vector_type(16))) _Float16 v16h;
typedef __attribute__((ext_vector_type(8)))  _Float16 v8h;
typedef __attribute__((ext_vector_type(8)))  float    v8f;

#define N_NODES 40000
#define N_EDGES 320000
#define GWIDTH  32
#define KIN     6
#define GDEPTH  4

// ---------------- small utility kernels ----------------

__global__ __launch_bounds__(256) void zero_f32(float* p, int n) {
    int i = blockIdx.x * 256 + threadIdx.x;
    if (i < n) p[i] = 0.0f;
}

__global__ __launch_bounds__(256) void zero_u32(unsigned* p, int n) {
    int i = blockIdx.x * 256 + threadIdx.x;
    if (i < n) p[i] = 0u;
}

__global__ __launch_bounds__(256) void deg_kernel(const int* ei, unsigned* cnt) {
    int e = blockIdx.x * 256 + threadIdx.x;
    if (e < N_EDGES) atomicAdd(&cnt[ei[N_EDGES + e]], 1u);
}

__global__ __launch_bounds__(256) void dinv_kernel(const unsigned* cnt, float* dinv) {
    int n = blockIdx.x * 256 + threadIdx.x;
    if (n < N_NODES) {
        unsigned c = cnt[n];
        dinv[n] = (c > 0u) ? (1.0f / (float)c) : 1.0f;
    }
}

// h0[n][j] = x[n]*fc1_w[j] + fc1_b[j]
__global__ __launch_bounds__(256) void init_h0(const float* x, const float* fc1w,
                                               const float* fc1b, float* h) {
    int idx = blockIdx.x * 256 + threadIdx.x;   // N*32 exact
    int n = idx >> 5, j = idx & 31;
    h[idx] = x[n] * fc1w[j] + fc1b[j];
}

// Pack a row-major f32 weight [K x (NT*16)] into WMMA B-fragment order (f16).
// Fragment layout per (ktile,ntile): 32 lanes x 16 halves, contiguous per lane.
// half index i<8  -> K = kt*32 + 8*(lane>=16) + i
// half index i>=8 -> K = kt*32 + 8*(lane>=16) + 16 + (i-8)
// column          -> nt*16 + (lane & 15)
__global__ __launch_bounds__(256) void pack_b(const float* src, _Float16* dst,
                                              int NT, int total) {
    int idx = blockIdx.x * 256 + threadIdx.x;
    if (idx >= total) return;
    int tile = idx >> 9;           // 512 halves per tile
    int within = idx & 511;
    int lane = within >> 4;
    int h = within & 15;
    int kt = tile / NT;
    int nt = tile % NT;
    int kc = kt * 32 + ((lane >> 4) << 3);
    int K = kc + ((h < 8) ? h : (h + 8));
    int n = nt * 16 + (lane & 15);
    dst[idx] = (_Float16)src[(size_t)K * (NT * 16) + n];
}

// ---------------- fused edge-MLP (WMMA) ----------------
// 4 waves / block, each wave owns 16 edges.
// layer1 (K=6) in VALU -> LDS f16 [16x128]; layer2/layer3 via v_wmma_f32_16x16x32_f16.
__global__ __launch_bounds__(128) void edge_mlp(const float* __restrict__ ea,
                                                const float* __restrict__ k1w,
                                                const float* __restrict__ k1b,
                                                const _Float16* __restrict__ k2p,
                                                const float* __restrict__ k2b,
                                                const _Float16* __restrict__ k3p,
                                                const float* __restrict__ k3b,
                                                _Float16* __restrict__ w16) {
    __shared__ alignas(16) _Float16 a1s[4][16 * 128];
    __shared__ alignas(16) _Float16 a2s[4][16 * 256];
    int wv = threadIdx.x >> 5;
    int lane = threadIdx.x & 31;
    int e0 = (blockIdx.x * 4 + wv) * 16;
    _Float16* a1 = a1s[wv];
    _Float16* a2 = a2s[wv];

    // ---- layer 1: [16,6] @ [6,128] + b, relu -> LDS f16 (wave-private) ----
    for (int idx = lane; idx < 16 * 128; idx += 32) {
        int m = idx >> 7, n = idx & 127;
        const float* ep = ea + (size_t)(e0 + m) * KIN;
        float s = k1b[n];
#pragma unroll
        for (int k = 0; k < KIN; ++k) s += ep[k] * k1w[k * 128 + n];
        a1[idx] = (_Float16)fmaxf(s, 0.0f);
    }

    int row  = lane & 15;
    int koff = (lane >> 4) << 3;     // +8 K-offset for lanes 16..31 (A/B frag layout)
    int mD   = (lane >> 4) << 3;     // D frag: row = r + 8*(lane>=16)
    int nD   = lane & 15;            // D frag: col = lane & 15

    // ---- layer 2: [16,128] @ [128,256] + b, relu -> LDS f16 ----
    for (int nt = 0; nt < 16; ++nt) {
        v8f c = {};
#pragma unroll
        for (int kt = 0; kt < 4; ++kt) {
            int kc = kt * 32 + koff;
            v8h lo = *(const v8h*)(a1 + row * 128 + kc);
            v8h hi = *(const v8h*)(a1 + row * 128 + kc + 16);
            v16h a;
#pragma unroll
            for (int i = 0; i < 8; ++i) { a[i] = lo[i]; a[8 + i] = hi[i]; }
            v16h b = *(const v16h*)(k2p + ((size_t)(kt * 16 + nt) * 32 + lane) * 16);
            c = __builtin_amdgcn_wmma_f32_16x16x32_f16(false, a, false, b,
                                                       (short)0, c, false, false);
        }
        float bias = k2b[nt * 16 + nD];
#pragma unroll
        for (int r = 0; r < 8; ++r)
            a2[(mD + r) * 256 + nt * 16 + nD] = (_Float16)fmaxf(c[r] + bias, 0.0f);
    }

    // ---- layer 3: [16,256] @ [256,1024] + b -> w16 (f16, global) ----
    v16h af[8];
#pragma unroll
    for (int kt = 0; kt < 8; ++kt) {
        int kc = kt * 32 + koff;
        v8h lo = *(const v8h*)(a2 + row * 256 + kc);
        v8h hi = *(const v8h*)(a2 + row * 256 + kc + 16);
#pragma unroll
        for (int i = 0; i < 8; ++i) { af[kt][i] = lo[i]; af[kt][8 + i] = hi[i]; }
    }
    for (int nt = 0; nt < 64; ++nt) {
        v8f c = {};
#pragma unroll
        for (int kt = 0; kt < 8; ++kt) {
            v16h b = *(const v16h*)(k3p + ((size_t)(kt * 64 + nt) * 32 + lane) * 16);
            c = __builtin_amdgcn_wmma_f32_16x16x32_f16(false, af[kt], false, b,
                                                       (short)0, c, false, false);
        }
        float bias = k3b[nt * 16 + nD];
        _Float16* wout = w16 + (size_t)e0 * 1024 + nt * 16 + nD;
#pragma unroll
        for (int r = 0; r < 8; ++r)
            wout[(size_t)(mD + r) * 1024] = (_Float16)(c[r] + bias);
    }
}

// ---------------- message passing ----------------
// One wave per edge: msg[o] = sum_i h[src][i] * w[e][i][o]; atomic scatter to dst.
__global__ __launch_bounds__(256) void msg_kernel(const float* __restrict__ hin,
                                                  const _Float16* __restrict__ w16,
                                                  const int* __restrict__ ei,
                                                  float* __restrict__ hacc) {
    int e = blockIdx.x * 8 + (threadIdx.x >> 5);
    int lane = threadIdx.x & 31;
    int src = ei[e];
    int dst = ei[N_EDGES + e];
    float hv = hin[(size_t)src * 32 + lane];
    const _Float16* wp = w16 + (size_t)e * 1024 + lane;
    float acc = 0.0f;
#pragma unroll
    for (int i = 0; i < 32; ++i) {
        float hi = __shfl(hv, i, 32);
        acc += hi * (float)wp[i * 32];
    }
    atomicAdd(&hacc[(size_t)dst * 32 + lane], acc);
}

__global__ __launch_bounds__(256) void norm_kernel(const float* hacc, const float* dinv,
                                                   float* hout, int do_relu) {
    int idx = blockIdx.x * 256 + threadIdx.x;   // N*32 exact
    float v = hacc[idx] * dinv[idx >> 5];
    hout[idx] = do_relu ? fmaxf(v, 0.0f) : v;
}

// ---------------- readout: relu(h@fc2+b) @ fc3 + b ----------------
__global__ __launch_bounds__(256) void readout(const float* __restrict__ h,
                                               const float* __restrict__ fc2w,
                                               const float* __restrict__ fc2b,
                                               const float* __restrict__ fc3w,
                                               const float* __restrict__ fc3b,
                                               float* __restrict__ out) {
    int n = blockIdx.x * 8 + (threadIdx.x >> 5);
    int lane = threadIdx.x & 31;
    float hv = h[(size_t)n * 32 + lane];
    float acc = 0.0f;
    for (int u0 = 0; u0 < 256; u0 += 32) {
        int u = u0 + lane;
        float s = fc2b[u];
#pragma unroll
        for (int j = 0; j < 32; ++j) {
            float hj = __shfl(hv, j, 32);
            s += hj * fc2w[j * 256 + u];
        }
        acc += fmaxf(s, 0.0f) * fc3w[u];
    }
#pragma unroll
    for (int off = 16; off; off >>= 1) acc += __shfl_down(acc, off, 32);
    if (lane == 0) out[n] = acc + fc3b[0];
}

// ---------------- launcher ----------------
extern "C" void kernel_launch(void* const* d_in, const int* in_sizes, int n_in,
                              void* d_out, int out_size, void* d_ws, size_t ws_size,
                              hipStream_t stream) {
    const float* x    = (const float*)d_in[0];
    const float* ea   = (const float*)d_in[1];
    const int*   ei   = (const int*)d_in[2];
    const float* fc1w = (const float*)d_in[3];
    const float* fc1b = (const float*)d_in[4];
    const float* k1w  = (const float*)d_in[5];
    const float* k1b  = (const float*)d_in[6];
    const float* k2w  = (const float*)d_in[7];
    const float* k2b  = (const float*)d_in[8];
    const float* k3w  = (const float*)d_in[9];
    const float* k3b  = (const float*)d_in[10];
    const float* fc2w = (const float*)d_in[11];
    const float* fc2b = (const float*)d_in[12];
    const float* fc3w = (const float*)d_in[13];
    const float* fc3b = (const float*)d_in[14];
    float* out = (float*)d_out;

    // workspace carve (all offsets 256B aligned)
    char* ws = (char*)d_ws;
    size_t off = 0;
    _Float16* w16 = (_Float16*)(ws + off); off += (size_t)N_EDGES * 1024 * 2;       // 655.36 MB
    float* h_a    = (float*)(ws + off);    off += (size_t)N_NODES * 32 * 4;         // 5.12 MB
    float* h_b    = (float*)(ws + off);    off += (size_t)N_NODES * 32 * 4;         // 5.12 MB
    float* dinv   = (float*)(ws + off);    off += (size_t)N_NODES * 4;
    unsigned* cnt = (unsigned*)(ws + off); off += (size_t)N_NODES * 4;
    _Float16* k2p = (_Float16*)(ws + off); off += (size_t)128 * 256 * 2;
    _Float16* k3p = (_Float16*)(ws + off); off += (size_t)256 * 1024 * 2;
    (void)ws_size; (void)in_sizes; (void)n_in; (void)out_size;

    // pack weights into WMMA B-fragment order
    pack_b<<<(128 * 256 + 255) / 256, 256, 0, stream>>>(k2w, k2p, 16, 128 * 256);
    pack_b<<<(256 * 1024 + 255) / 256, 256, 0, stream>>>(k3w, k3p, 64, 256 * 1024);

    // degrees -> 1/denom
    zero_u32<<<(N_NODES + 255) / 256, 256, 0, stream>>>(cnt, N_NODES);
    deg_kernel<<<(N_EDGES + 255) / 256, 256, 0, stream>>>(ei, cnt);
    dinv_kernel<<<(N_NODES + 255) / 256, 256, 0, stream>>>(cnt, dinv);

    // h0 = x @ fc1 + b
    init_h0<<<N_NODES * 32 / 256, 256, 0, stream>>>(x, fc1w, fc1b, h_a);

    // per-edge 32x32 weight matrices via WMMA, stored f16
    edge_mlp<<<N_EDGES / 64, 128, 0, stream>>>(ea, k1w, k1b, k2p, k2b, k3p, k3b, w16);

    // 4 rounds of gather-matvec-scatter-mean
    for (int k = 0; k < GDEPTH; ++k) {
        zero_f32<<<N_NODES * 32 / 256, 256, 0, stream>>>(h_b, N_NODES * 32);
        msg_kernel<<<N_EDGES / 8, 256, 0, stream>>>(h_a, w16, ei, h_b);
        norm_kernel<<<N_NODES * 32 / 256, 256, 0, stream>>>(h_b, dinv, h_a,
                                                            (k != GDEPTH - 1) ? 1 : 0);
    }

    // readout
    readout<<<N_NODES / 8, 256, 0, stream>>>(h_a, fc2w, fc2b, fc3w, fc3b, out);
}
